// Global_ST_70317204570407
// MI455X (gfx1250) — matrix operand order repaired and checked by reading, not compiled, not run on hardware
//
#include <hip/hip_runtime.h>
#include <hip/hip_bf16.h>

// ---------------------------------------------------------------------------
// GatedBTCN + AGCN, fused for MI455X (gfx1250, wave32, WMMA bf16).
//
//  * Conv stack + BNs algebraically folded into composite weights (prep kernels).
//  * One workgroup (256 thr = 8 waves) per (b, t) slice:
//      phase 1 (4 row-passes of 128): X[128x256]·Wc -> gate -> xg·W2 -> Y (LDS bf16)
//      phase 2: Out = relu( (At[512x512]·Y[512x64]) * asc + ash )
//  * v_wmma_f32_16x16x32_bf16 everywhere; weights/At pre-shuffled so operand
//    loads are contiguous b128 per lane.
//  * sched_group_barrier pins per-iteration order: ALL fragment loads first
//    (one deep clause, latency overlapped), then cvts, then WMMAs; sched_barrier
//    at iteration ends bounds liveness (<=256 VGPRs, no spills, 4 waves/SIMD).
// ---------------------------------------------------------------------------

#define EPSF 1e-5f

typedef __attribute__((ext_vector_type(16))) __bf16      v16bf;
typedef __attribute__((ext_vector_type(8)))  float       v8f;
typedef __attribute__((ext_vector_type(4)))  float       v4f;
typedef __attribute__((ext_vector_type(4)))  unsigned int v4u;

union Frag {
  v16bf v;
  unsigned short u[16];
  v4u q[2];
};

__device__ __forceinline__ unsigned short bf16bits(float x) {
  return __builtin_bit_cast(unsigned short, (__bf16)x);
}

// sigmoid via v_exp + v_rcp only (no IEEE divide chain)
__device__ __forceinline__ float fast_sigmoid(float x) {
  return __builtin_amdgcn_rcpf(1.0f + __expf(-x));
}
__device__ __forceinline__ float fast_tanh(float x) {
  return 2.0f * fast_sigmoid(2.0f * x) - 1.0f;
}

// hard scheduling fence: nothing moves across this point
__device__ __forceinline__ void sched_fence() {
  __builtin_amdgcn_sched_barrier(0);
}
// scheduling groups (masks: 0x2 VALU, 0x8 MFMA/WMMA, 0x20 VMEM read, 0x100 DS read)
#define SGB(mask, n) __builtin_amdgcn_sched_group_barrier((mask), (n), 0)

// ---------------------------------------------------------------------------
// Prep kernel 1: At_bf16[m][n] = bf16(A[n][m])   (A-operand of phase-2 GEMM)
// ---------------------------------------------------------------------------
__global__ void prep_at(const float* __restrict__ A, unsigned short* __restrict__ At) {
  int idx = blockIdx.x * 256 + threadIdx.x;
  if (idx >= 512 * 512) return;
  int m = idx >> 9, n = idx & 511;
  At[m * 512 + n] = bf16bits(A[n * 512 + m]);
}

// ---------------------------------------------------------------------------
// Prep kernel 2: composite gate weights, fragment-shuffled.
//   Wc[sigma][g][k=tb*64+cin][f] = sum_c gw[f,c,kh] * sbn[c] * bcw_br[c,cin,j]
//   stored bf16 [sigma][kb(8)][ct(8)][lane(32)][e(16)]  (B-fragment order)
// ---------------------------------------------------------------------------
__global__ void prep_wc(const float* __restrict__ bc1w, const float* __restrict__ bc2w,
                        const float* __restrict__ bbng, const float* __restrict__ bbnv,
                        const float* __restrict__ gc1w, const float* __restrict__ gc2w,
                        unsigned short* __restrict__ Wc) {
  int idx = blockIdx.x * 256 + threadIdx.x;
  if (idx >= 3 * 2 * 256 * 64) return;
  int f  = idx & 63;
  int k  = (idx >> 6) & 255;
  int g  = (idx >> 14) & 1;
  int sg = idx >> 15;                 // sigma: 0,1,2
  int tb = k >> 6, cin = k & 63;
  int kh = tb >> 1, j = tb & 1;
  int br = (sg == 0) ? 0 : ((sg == 2) ? 1 : kh);
  const float* bw = br ? bc2w : bc1w;       // [CH=32][F=64][2]
  const float* gw = g  ? gc2w : gc1w;       // [F=64][CH=32][2]
  float acc = 0.f;
  for (int c = 0; c < 32; ++c) {
    float s = bbng[c] * rsqrtf(bbnv[c] + EPSF);
    acc += gw[(f * 32 + c) * 2 + kh] * s * bw[(c * 64 + cin) * 2 + j];
  }
  int col  = g * 64 + f;
  int kb   = k >> 5;
  int ct   = col >> 4;
  int lane = (col & 15) + (((k >> 4) & 1) << 4);
  int e    = k & 15;
  Wc[(((sg * 8 + kb) * 8 + ct) * 32 + lane) * 16 + e] = bf16bits(acc);
}

// ---------------------------------------------------------------------------
// Prep kernel 3: W2 shuffle + composite gate biases + epilogue vectors.
// ---------------------------------------------------------------------------
__global__ void prep_small(const float* __restrict__ Wm,
                           const float* __restrict__ gbng, const float* __restrict__ gbnb,
                           const float* __restrict__ gbnm, const float* __restrict__ gbnv,
                           const float* __restrict__ gc1w, const float* __restrict__ gc1b,
                           const float* __restrict__ gc2w, const float* __restrict__ gc2b,
                           const float* __restrict__ bc1b, const float* __restrict__ bc2b,
                           const float* __restrict__ bbng, const float* __restrict__ bbnb,
                           const float* __restrict__ bbnm, const float* __restrict__ bbnv,
                           const float* __restrict__ abng, const float* __restrict__ abnb,
                           const float* __restrict__ abnm, const float* __restrict__ abnv,
                           unsigned short* __restrict__ W2s, float* __restrict__ Bc,
                           float* __restrict__ b2, float* __restrict__ ascv,
                           float* __restrict__ ashv) {
  int idx = blockIdx.x * 256 + threadIdx.x;
  if (idx < 4096) {
    // W2[c][f] = sg_c * W[c][f], fragment-shuffled  [kb(2)][ct(4)][lane(32)][e(16)]
    int c = idx >> 6, f = idx & 63;
    float s = gbng[c] * rsqrtf(gbnv[c] + EPSF);
    float v = s * Wm[c * 64 + f];
    int kb = c >> 5, ct = f >> 4;
    int lane = (f & 15) + (((c >> 4) & 1) << 4);
    int e = c & 15;
    W2s[((kb * 4 + ct) * 32 + lane) * 16 + e] = bf16bits(v);
  } else if (idx < 4096 + 384) {
    // Bc[sigma][g][f]: composite gate bias
    int i = idx - 4096;
    int f = i & 63, g = (i >> 6) & 1, sg = i >> 7;
    const float* gw = g ? gc2w : gc1w;
    const float* gb = g ? gc2b : gc1b;
    float acc = gb[f];
    for (int kh = 0; kh < 2; ++kh) {
      int br = (sg == 0) ? 0 : ((sg == 2) ? 1 : kh);
      const float* bb = br ? bc2b : bc1b;
      for (int c = 0; c < 32; ++c) {
        float s = bbng[c] * rsqrtf(bbnv[c] + EPSF);
        float o = bbnb[c] - bbnm[c] * s;
        acc += gw[(f * 32 + c) * 2 + kh] * (bb[c] * s + o);
      }
    }
    Bc[i] = acc;
  } else if (idx < 4096 + 384 + 64) {
    int f = idx - 4480;
    float accb = 0.f;
    for (int c = 0; c < 64; ++c) {
      float s = gbng[c] * rsqrtf(gbnv[c] + EPSF);
      float o = gbnb[c] - gbnm[c] * s;
      accb += o * Wm[c * 64 + f];
    }
    b2[f] = accb;
    float sa = abng[f] * rsqrtf(abnv[f] + EPSF);
    ascv[f] = sa;
    ashv[f] = abnb[f] - abnm[f] * sa;
  }
}

// ---------------------------------------------------------------------------
// Main fused kernel: one workgroup per (b, t) slice.
// ---------------------------------------------------------------------------
#define YSTRIDE 68   // ybuf row stride (bf16 elems): 16-lane halves 32 banks apart
#define XSTRIDE 72   // xgbuf row stride: 144 B rows -> 16 B aligned for b128 reads

__global__ __launch_bounds__(256, 4) void fused_gbtcn_agcn(
    const float* __restrict__ X,            // [32][48][512][64] f32
    const unsigned short* __restrict__ At,  // bf16 [512][512]  (= A^T)
    const unsigned short* __restrict__ Wc,  // bf16 shuffled [3][8][8][32][16]
    const unsigned short* __restrict__ W2s, // bf16 shuffled [2][4][32][16]
    const float* __restrict__ Bc,           // [3][2][64]
    const float* __restrict__ b2,           // [64]
    const float* __restrict__ ascv,         // [64]
    const float* __restrict__ ashv,         // [64]
    float* __restrict__ Out)                // [32][46][512][64] f32
{
  __shared__ unsigned short ybuf[512 * YSTRIDE];   // 69632 B : Y slice (bf16)
  __shared__ unsigned short xgbuf[128 * XSTRIDE];  // 18432 B : per-wave xg staging

  const int t = blockIdx.x;          // 0..45
  const int b = blockIdx.y;          // 0..31
  const int tid = threadIdx.x;
  const int wave = tid >> 5;
  const int lane = tid & 31;
  const int l16 = lane & 15;
  const int lhi = lane >> 4;         // 0 or 1

  // time taps of the composed conv (handles h1|h2 concat boundary)
  const int tt0 = (t < 24) ? t : (t - 24);
  const int tt1 = (t + 2 < 24) ? (t + 2) : (t - 22);
  const int sigma = (t < 22) ? 0 : ((t < 24) ? 1 : 2);
  int srcT[4];
  srcT[0] = 2 * tt0;     srcT[1] = 2 * tt0 + 1;
  srcT[2] = 2 * tt1;     srcT[3] = 2 * tt1 + 1;

  const float* Xb = X + (size_t)b * 48 * 512 * 64;

  // ============ Phase 1: build Y slice (4 row-passes of 128 rows) =============
  #pragma unroll 1
  for (int pass = 0; pass < 4; ++pass) {
    const int rowbase = pass * 128 + wave * 16;   // node index base for this wave

    // ---- GEMM1: gates = Xrow[16x256] . Wc[256x128] ----
    v8f acc[8];
    #pragma unroll
    for (int ct = 0; ct < 8; ++ct)
      #pragma unroll
      for (int r = 0; r < 8; ++r) acc[ct][r] = 0.f;

    #pragma unroll 1
    for (int kb = 0; kb < 8; ++kb) {
      const int tb = kb >> 1;
      const int cinb = (kb & 1) << 5;
      const float* xsrc = Xb + (size_t)srcT[tb] * 512 * 64;
      // batch: A-source f32 loads + all 8 B-fragment loads issue together
      Frag af;
      Frag bf[8];
      const float* p = xsrc + (size_t)(rowbase + l16) * 64 + cinb + lhi * 16;
      v4f f0 = *(const v4f*)(p);
      v4f f1 = *(const v4f*)(p + 4);
      v4f f2 = *(const v4f*)(p + 8);
      v4f f3 = *(const v4f*)(p + 12);
      const unsigned short* wcp = Wc + (size_t)((sigma * 8 + kb) * 8) * 512;
      #pragma unroll
      for (int ct = 0; ct < 8; ++ct) {
        const unsigned short* q = wcp + (ct * 32 + lane) * 16;
        bf[ct].q[0] = *(const v4u*)(q);
        bf[ct].q[1] = *(const v4u*)(q + 8);
      }
      #pragma unroll
      for (int i = 0; i < 4; ++i) {
        af.u[i]      = bf16bits(f0[i]);
        af.u[4 + i]  = bf16bits(f1[i]);
        af.u[8 + i]  = bf16bits(f2[i]);
        af.u[12 + i] = bf16bits(f3[i]);
      }
      #pragma unroll
      for (int ct = 0; ct < 8; ++ct)
        acc[ct] = __builtin_amdgcn_wmma_f32_16x16x32_bf16(
            false, af.v, false, bf[ct].v, (short)0, acc[ct], false, false);
      // pinned order: all 20 vmem reads -> cvts -> 8 wmma (forces live fragments)
      SGB(0x020, 20);   // VMEM reads (4 X + 16 Wc)
      SGB(0x002, 16);   // VALU (bf16 cvts)
      SGB(0x008, 8);    // WMMA
      sched_fence();    // region delimiter: no cross-iteration hoisting
    }

    // ---- gating: xg = tanh(g1) * sigmoid(g2), bf16 into per-wave LDS slab ----
    #pragma unroll
    for (int j = 0; j < 4; ++j) {
      const float b1v = Bc[(sigma * 2 + 0) * 64 + j * 16 + l16];
      const float b2v = Bc[(sigma * 2 + 1) * 64 + j * 16 + l16];
      #pragma unroll
      for (int r = 0; r < 8; ++r) {
        float g1 = acc[j][r] + b1v;
        float g2 = acc[j + 4][r] + b2v;
        float xg = fast_tanh(g1) * fast_sigmoid(g2);
        int lrow = wave * 16 + r + lhi * 8;
        xgbuf[lrow * XSTRIDE + j * 16 + l16] = bf16bits(xg);
      }
    }
    sched_fence();
    // same-wave LDS RAW: compiler inserts s_wait_dscnt; rows are wave-private.

    // ---- GEMM2: Y = xg[16x64] . W2[64x64] + bias2  -> ybuf ----
    v8f acc2[4];
    #pragma unroll
    for (int nt = 0; nt < 4; ++nt)
      #pragma unroll
      for (int r = 0; r < 8; ++r) acc2[nt][r] = 0.f;

    #pragma unroll 1
    for (int kb = 0; kb < 2; ++kb) {
      Frag a2;
      Frag bf[4];
      {
        const unsigned short* p =
            &xgbuf[(wave * 16 + l16) * XSTRIDE + kb * 32 + lhi * 16];
        a2.q[0] = *(const v4u*)(p);
        a2.q[1] = *(const v4u*)(p + 8);
      }
      #pragma unroll
      for (int nt = 0; nt < 4; ++nt) {
        const unsigned short* q = W2s + ((kb * 4 + nt) * 32 + lane) * 16;
        bf[nt].q[0] = *(const v4u*)(q);
        bf[nt].q[1] = *(const v4u*)(q + 8);
      }
      #pragma unroll
      for (int nt = 0; nt < 4; ++nt)
        acc2[nt] = __builtin_amdgcn_wmma_f32_16x16x32_bf16(
            false, a2.v, false, bf[nt].v, (short)0, acc2[nt], false, false);
      SGB(0x100, 2);    // DS reads (xg A-fragment)
      SGB(0x020, 8);    // VMEM reads (W2 fragments)
      SGB(0x008, 4);    // WMMA
      sched_fence();
    }
    #pragma unroll
    for (int nt = 0; nt < 4; ++nt) {
      const float bb = b2[nt * 16 + l16];
      #pragma unroll
      for (int r = 0; r < 8; ++r) {
        int nrow = rowbase + r + lhi * 8;
        ybuf[nrow * YSTRIDE + nt * 16 + l16] = bf16bits(acc2[nt][r] + bb);
      }
    }
    sched_fence();   // keep pass iterations from overlapping register live ranges
  }

  __syncthreads();

  // ============ Phase 2: Out = relu( (At . Y) * asc + ash ) ===================
  const int mg = wave >> 1;   // 0..3 : 128-row block of output m
  const int fg = wave & 1;    // 0..1 : 32-col block of f
  v8f acc3[8][2];
  #pragma unroll
  for (int mt = 0; mt < 8; ++mt)
    #pragma unroll
    for (int j = 0; j < 2; ++j)
      #pragma unroll
      for (int r = 0; r < 8; ++r) acc3[mt][j][r] = 0.f;

  #pragma unroll 1
  for (int kb = 0; kb < 16; ++kb) {
    Frag bfr[2];
    #pragma unroll
    for (int j = 0; j < 2; ++j) {
      const int f = (fg * 2 + j) * 16 + l16;
      const int nb = kb * 32 + lhi * 16;
      #pragma unroll
      for (int e = 0; e < 16; ++e)
        bfr[j].u[e] = ybuf[(nb + e) * YSTRIDE + f];
    }
    // two chunks of 4 m-tiles: batch 4 At fragments, then 8 WMMAs
    #pragma unroll
    for (int half = 0; half < 2; ++half) {
      Frag afr[4];
      #pragma unroll
      for (int q4 = 0; q4 < 4; ++q4) {
        const int mt = half * 4 + q4;
        const unsigned short* p =
            At + (size_t)(mg * 128 + mt * 16 + l16) * 512 + kb * 32 + lhi * 16;
        afr[q4].q[0] = *(const v4u*)(p);
        afr[q4].q[1] = *(const v4u*)(p + 8);
      }
      #pragma unroll
      for (int q4 = 0; q4 < 4; ++q4) {
        const int mt = half * 4 + q4;
        #pragma unroll
        for (int j = 0; j < 2; ++j)
          acc3[mt][j] = __builtin_amdgcn_wmma_f32_16x16x32_bf16(
              false, afr[q4].v, false, bfr[j].v, (short)0, acc3[mt][j], false, false);
      }
    }
    // pinned order: Y gathers, then (8 At loads -> 8 wmma) per half
    SGB(0x100, 32);   // DS reads (Y fragment gathers)
    SGB(0x020, 8);    // VMEM reads (At half 0)
    SGB(0x008, 8);    // WMMA half 0
    SGB(0x020, 8);    // VMEM reads (At half 1)
    SGB(0x008, 8);    // WMMA half 1
    sched_fence();    // bound liveness: no cross-k hoisting
  }

  float* outp = Out + (size_t)(b * 46 + t) * 512 * 64;
  #pragma unroll
  for (int j = 0; j < 2; ++j) {
    const int f = (fg * 2 + j) * 16 + l16;
    const float s = ascv[f];
    const float sh = ashv[f];
    #pragma unroll
    for (int mt = 0; mt < 8; ++mt) {
      #pragma unroll
      for (int r = 0; r < 8; ++r) {
        const int m = mg * 128 + mt * 16 + r + lhi * 8;
        float v = acc3[mt][j][r] * s + sh;
        outp[(size_t)m * 64 + f] = (v > 0.f) ? v : 0.f;
      }
    }
  }
}

// ---------------------------------------------------------------------------
// Host launcher
// ---------------------------------------------------------------------------
extern "C" void kernel_launch(void* const* d_in, const int* in_sizes, int n_in,
                              void* d_out, int out_size, void* d_ws, size_t ws_size,
                              hipStream_t stream) {
  const float* X    = (const float*)d_in[0];
  const float* A    = (const float*)d_in[1];
  const float* bc1w = (const float*)d_in[2];
  const float* bc1b = (const float*)d_in[3];
  const float* bc2w = (const float*)d_in[4];
  const float* bc2b = (const float*)d_in[5];
  const float* bbng = (const float*)d_in[6];
  const float* bbnb = (const float*)d_in[7];
  const float* bbnm = (const float*)d_in[8];
  const float* bbnv = (const float*)d_in[9];
  const float* gc1w = (const float*)d_in[10];
  const float* gc1b = (const float*)d_in[11];
  const float* gc2w = (const float*)d_in[12];
  const float* gc2b = (const float*)d_in[13];
  const float* gbng = (const float*)d_in[14];
  const float* gbnb = (const float*)d_in[15];
  const float* gbnm = (const float*)d_in[16];
  const float* gbnv = (const float*)d_in[17];
  const float* Wm   = (const float*)d_in[18];
  const float* abng = (const float*)d_in[19];
  const float* abnb = (const float*)d_in[20];
  const float* abnm = (const float*)d_in[21];
  const float* abnv = (const float*)d_in[22];

  char* ws = (char*)d_ws;
  unsigned short* At  = (unsigned short*)(ws + 0);        // 524288 B
  unsigned short* Wc  = (unsigned short*)(ws + 524288);   // 196608 B
  unsigned short* W2s = (unsigned short*)(ws + 720896);   //   8192 B
  float* Bc   = (float*)(ws + 729088);                    //   1536 B
  float* b2   = (float*)(ws + 730624);                    //    256 B
  float* ascv = (float*)(ws + 730880);                    //    256 B
  float* ashv = (float*)(ws + 731136);                    //    256 B

  prep_at<<<1024, 256, 0, stream>>>(A, At);
  prep_wc<<<384, 256, 0, stream>>>(bc1w, bc2w, bbng, bbnv, gc1w, gc2w, Wc);
  prep_small<<<18, 256, 0, stream>>>(Wm, gbng, gbnb, gbnm, gbnv,
                                     gc1w, gc1b, gc2w, gc2b,
                                     bc1b, bc2b, bbng, bbnb, bbnm, bbnv,
                                     abng, abnb, abnm, abnv,
                                     W2s, Bc, b2, ascv, ashv);

  dim3 grid(46, 32);
  fused_gbtcn_agcn<<<grid, 256, 0, stream>>>(X, At, Wc, W2s, Bc, b2, ascv, ashv,
                                             (float*)d_out);
}
